// SimpleConvWithPooling_6794638262524
// MI455X (gfx1250) — matrix operand order repaired and checked by reading, not compiled, not run on hardware
//
#include <hip/hip_runtime.h>
#include <hip/hip_bf16.h>
#include <math.h>

typedef __attribute__((ext_vector_type(16))) _Float16 v16h;
typedef __attribute__((ext_vector_type(8)))  float    v8f;
typedef __attribute__((ext_vector_type(4)))  float    v4f;

#define BATCHES 65536
#define LEN 1024
#define KW 16
#define LOUT 1009              // 1024 - 16 + 1
#define WAVES_PER_BLOCK 8

__device__ __forceinline__ int imin_i(int a, int b) { return a < b ? a : b; }

__global__ __launch_bounds__(256)
void conv_pool_classify(const float* __restrict__ x,
                        const float* __restrict__ w,
                        const float* __restrict__ bptr,
                        const float* __restrict__ ranges,
                        float* __restrict__ out)
{
    const int lane  = threadIdx.x & 31;
    const int wave  = threadIdx.x >> 5;
    const int batch = blockIdx.x * WAVES_PER_BLOCK + wave;   // one wave per batch
    const float* xb = x + (size_t)batch * LEN;

    // ---- Build B fragment: 32x16 Toeplitz matrix of w, B[kk][n] = w[kk-n] ----
    // 16-bit B layout (wave32): lane<16 holds col n=lane, rows 0..7 & 16..23 (2/VGPR);
    // lane>=16 holds col n=lane-16, rows 8..15 & 24..31.
    _Float16 wh[KW];
#pragma unroll
    for (int i = 0; i < KW; ++i) wh[i] = (_Float16)w[i];

    const int ncol    = lane & 15;
    const int rowBase = (lane >> 4) << 4;                    // 0 or 16
    v16h Bfrag;
#pragma unroll
    for (int j = 0; j < 16; ++j) {
        const int kk = (j < 8 ? 0 : 8) + (rowBase >> 1) + j;
        const int d  = kk - ncol;
        Bfrag[j] = (d >= 0 && d < KW) ? wh[d] : (_Float16)0.0f;
    }

    const float bias = bptr[0];
    const int r    = lane & 15;
    const int h8   = (lane >> 4) << 3;                       // 0 or 8
    const int mAdd = (lane >> 4) << 3;                       // D: M = rr + 8*(lane>=16)

    float vmax = 0.0f;                                       // ReLU => valid max >= 0

#pragma unroll
    for (int tile = 0; tile < 4; ++tile) {
        const int base = tile * 256;
        // 16-bit A layout: lane<16 holds row m=r, K=0..7 & 16..23;
        // lane>=16 holds row m=r, K=8..15 & 24..31  =>  per-lane base i0:
        const int i0 = base + 16 * r + h8;
        // Clamp chunk starts so tile 3 never reads past x[1023]; clamped chunks
        // only feed outputs with t > 1008 which are masked below.
        const int c0 = imin_i(i0,      LEN - 4);
        const int c1 = imin_i(i0 + 4,  LEN - 4);
        const int c2 = imin_i(i0 + 16, LEN - 4);
        const int c3 = imin_i(i0 + 20, LEN - 4);
        const v4f a0 = *(const v4f*)(xb + c0);
        const v4f a1 = *(const v4f*)(xb + c1);
        const v4f a2 = *(const v4f*)(xb + c2);
        const v4f a3 = *(const v4f*)(xb + c3);

        v16h A;
#pragma unroll
        for (int j = 0; j < 4; ++j) {
            A[j]      = (_Float16)a0[j];
            A[4 + j]  = (_Float16)a1[j];
            A[8 + j]  = (_Float16)a2[j];
            A[12 + j] = (_Float16)a3[j];
        }

        v8f acc = {};
        // D[m][n] = sum_kk x[base+16m+kk] * w[kk-n] = y_conv[base + 16m + n]
        acc = __builtin_amdgcn_wmma_f32_16x16x32_f16(false, A, false, Bfrag,
                                                     (short)0, acc, false, false);

#pragma unroll
        for (int rr = 0; rr < 8; ++rr) {
            const int M = rr + mAdd;
            const int t = base + 16 * M + ncol;
            float y = acc[rr] + bias;
            y = fmaxf(y, 0.0f);
            y = (t < LOUT) ? y : 0.0f;                        // branchless mask
            vmax = fmaxf(vmax, y);
        }
    }

    // ---- wave32 max reduction ----
#pragma unroll
    for (int off = 16; off > 0; off >>= 1)
        vmax = fmaxf(vmax, __shfl_xor(vmax, off, 32));

    // ---- classification (lane 0 only, after all WMMAs) ----
    if (lane == 0) {
        const float v = vmax;
        int label = -1;
#pragma unroll
        for (int i = 0; i < 4; ++i) {
            const float lo = ranges[2 * i];
            const float hi = ranges[2 * i + 1];
            if (label < 0 && v >= lo && v <= hi) label = i;   // first match
        }
        if (label < 0) {
            float best = __builtin_inff();
            int   bi   = 0;
#pragma unroll
            for (int i = 0; i < 8; ++i) {
                const float d = sqrtf(v * v + ranges[i]);     // flat bounds
                if (d < best) { best = d; bi = i; }           // first argmin
            }
            label = bi >> 1;
        }
        out[batch] = (float)label;
    }
}

extern "C" void kernel_launch(void* const* d_in, const int* in_sizes, int n_in,
                              void* d_out, int out_size, void* d_ws, size_t ws_size,
                              hipStream_t stream) {
    const float* x      = (const float*)d_in[0];   // (65536, 1, 1024) f32
    const float* w      = (const float*)d_in[1];   // (1, 1, 16) f32
    const float* b      = (const float*)d_in[2];   // (1,) f32
    const float* ranges = (const float*)d_in[3];   // (4, 2) f32
    float* out          = (float*)d_out;           // 65536 labels (as float)

    dim3 grid(BATCHES / WAVES_PER_BLOCK);          // 8192 blocks x 256 threads (8 waves)
    conv_pool_classify<<<grid, 256, 0, stream>>>(x, w, b, ranges, out);
}
